// GIN_74062416053220
// MI455X (gfx1250) — matrix operand order repaired and checked by reading, not compiled, not run on hardware
//
#include <hip/hip_runtime.h>
#include <hip/hip_bf16.h>

#define DIM     64
#define LAYERS  3
#define CLASSES 10
#define GRAPHS  512
#define BN_EPSF 1e-5f

typedef _Float16 v16h __attribute__((ext_vector_type(16)));
typedef float    v8f  __attribute__((ext_vector_type(8)));

// ---------------------------------------------------------------- utilities
__global__ void gin_zero_f32(float* __restrict__ p, int n) {
  int i = blockIdx.x * blockDim.x + threadIdx.x;
  for (; i < n; i += gridDim.x * blockDim.x) p[i] = 0.0f;
}

__global__ void gin_copy_f32v4(float4* __restrict__ dst, const float4* __restrict__ src, long n4) {
  long i = blockIdx.x * (long)blockDim.x + threadIdx.x;
  long step = (long)gridDim.x * blockDim.x;
  for (; i < n4; i += step) dst[i] = src[i];
}

// ------------------------------------------------- weight pre-pack: f32 -> f16 B-fragments
// wpack layout per matrix: [tile(4)][ks(2)][lane(32)][16 halfs] == 4096 halfs (one 64x64 mat)
// matrices ordered: l0.w1, l0.w2, l1.w1, l1.w2, l2.w1, l2.w2
__global__ void gin_pack_weights(const float* __restrict__ w1, const float* __restrict__ w2,
                                 _Float16* __restrict__ wpack) {
  int t = blockIdx.x * blockDim.x + threadIdx.x;
  const int total = LAYERS * 2 * 4096;
  if (t >= total) return;
  int j     = t & 15;
  int lane  = (t >> 4) & 31;
  int ks    = (t >> 9) & 1;
  int tile  = (t >> 10) & 3;
  int m2    = t >> 12;              // 0..2L-1
  int l     = m2 >> 1;
  int which = m2 & 1;
  int n     = lane & 15;
  int kb16  = (lane >> 4) * 16;
  int k     = ks * 32 + (kb16 >> 1) + (j & 7) + ((j >> 3) << 4);
  int col   = tile * 16 + n;
  const float* W = (which ? w2 : w1) + (size_t)l * DIM * DIM;
  wpack[t] = (_Float16)W[k * DIM + col];
}

// ------------------------------------------------- edge scatter: agg[dst] += h[src]
__global__ void gin_edge_scatter(float* __restrict__ agg, const float* __restrict__ h,
                                 const int* __restrict__ src, const int* __restrict__ dst,
                                 int n_edges) {
  long t = blockIdx.x * (long)blockDim.x + threadIdx.x;
  long total = (long)n_edges * (DIM / 4);
  long step = (long)gridDim.x * blockDim.x;
  for (; t < total; t += step) {
    int e = (int)(t >> 4);
    int p = (int)(t & 15);
    int s = src[e];
    int d = dst[e];
    float4 v = *(const float4*)(h + (long)s * DIM + p * 4);
    float* base = agg + (long)d * DIM + p * 4;
    atomicAdd(base + 0, v.x);
    atomicAdd(base + 1, v.y);
    atomicAdd(base + 2, v.z);
    atomicAdd(base + 3, v.w);
  }
}

// ------------------------------------------------- fused 2-layer MLP via WMMA (in-place)
// z = relu(relu(z @ W1 + b1) @ W2 + b2); block = 16 rows x 64 cols, 4 wave32s
__global__ __launch_bounds__(128)
void gin_mlp_wmma(float* __restrict__ z,
                  const _Float16* __restrict__ W1p, const float* __restrict__ B1,
                  const _Float16* __restrict__ W2p, const float* __restrict__ B2,
                  int n_nodes) {
  __shared__ _Float16 z1s[16][72];    // inter-GEMM f16 activations (padded stride)
  __shared__ float    outs[16][64];   // f32 output tile for coalesced writeback

  const int row0  = blockIdx.x * 16;
  const int wave  = threadIdx.x >> 5;
  const int lane  = threadIdx.x & 31;
  const int ncol0 = wave * 16;

  // CDNA5 fragment index helpers
  const int m     = lane & 15;          // A-frag row
  const int klo   = (lane >> 4) * 8;    // A-frag K sub-offset
  const int n     = lane & 15;          // B/C-frag column
  const int mbase = (lane >> 4) * 8;    // C-frag row base

  int arow_idx = row0 + m;
  if (arow_idx >= n_nodes) arow_idx = n_nodes - 1;   // keep EXEC all-ones for WMMA
  const float* arow = z + (long)arow_idx * DIM;

  // ---------------- GEMM 1: z1 = relu(z @ W1 + b1) ----------------
  v8f c1 = {};
#pragma unroll
  for (int ks = 0; ks < 2; ++ks) {
    const int kbase = ks * 32;
    // A-frag: two contiguous 8-float runs -> 4x b128 loads, pk converts
    float4 a0 = *(const float4*)(arow + kbase + klo);
    float4 a1 = *(const float4*)(arow + kbase + klo + 4);
    float4 a2 = *(const float4*)(arow + kbase + 16 + klo);
    float4 a3 = *(const float4*)(arow + kbase + 16 + klo + 4);
    v16h a;
    a[0]  = (_Float16)a0.x; a[1]  = (_Float16)a0.y; a[2]  = (_Float16)a0.z; a[3]  = (_Float16)a0.w;
    a[4]  = (_Float16)a1.x; a[5]  = (_Float16)a1.y; a[6]  = (_Float16)a1.z; a[7]  = (_Float16)a1.w;
    a[8]  = (_Float16)a2.x; a[9]  = (_Float16)a2.y; a[10] = (_Float16)a2.z; a[11] = (_Float16)a2.w;
    a[12] = (_Float16)a3.x; a[13] = (_Float16)a3.y; a[14] = (_Float16)a3.z; a[15] = (_Float16)a3.w;
    // B-frag: single contiguous 32B per lane (pre-packed f16)
    v16h b = *(const v16h*)(W1p + (((wave * 2 + ks) * 32 + lane) << 4));
    c1 = __builtin_amdgcn_wmma_f32_16x16x32_f16(false, a, false, b, (short)0, c1, false, false);
  }
  {
    const float bias = B1[ncol0 + n];
#pragma unroll
    for (int v = 0; v < 8; ++v) {
      float t = c1[v] + bias;
      t = t > 0.0f ? t : 0.0f;
      z1s[mbase + v][ncol0 + n] = (_Float16)t;
    }
  }
  __syncthreads();   // all stage-1 global reads done before any in-place store later

  // ---------------- GEMM 2: z = relu(z1 @ W2 + b2) ----------------
  v8f c2 = {};
#pragma unroll
  for (int ks = 0; ks < 2; ++ks) {
    const int kbase = ks * 32;
    v16h a, b;
#pragma unroll
    for (int j = 0; j < 8; ++j) {
      a[j]     = z1s[m][kbase + klo + j];
      a[8 + j] = z1s[m][kbase + 16 + klo + j];
    }
    b = *(const v16h*)(W2p + (((wave * 2 + ks) * 32 + lane) << 4));
    c2 = __builtin_amdgcn_wmma_f32_16x16x32_f16(false, a, false, b, (short)0, c2, false, false);
  }
  {
    const float bias = B2[ncol0 + n];
#pragma unroll
    for (int v = 0; v < 8; ++v) {
      float t = c2[v] + bias;
      outs[mbase + v][ncol0 + n] = t > 0.0f ? t : 0.0f;
    }
  }
  __syncthreads();

  // coalesced writeback: 256 float4 slots across 128 threads
#pragma unroll
  for (int i = 0; i < 2; ++i) {
    int idx = threadIdx.x * 2 + i;
    int r   = idx >> 4;
    int cc  = (idx & 15) * 4;
    int gr  = row0 + r;
    if (gr < n_nodes)
      *(float4*)(z + (long)gr * DIM + cc) = *(const float4*)&outs[r][cc];
  }
}

// ------------------------------------------------- BatchNorm pass 1: column sums
__global__ void gin_bn_stats(const float* __restrict__ z, float* __restrict__ stats,
                             int n_nodes) {
  const int c = threadIdx.x & 63;
  const int rows_per_blk = blockDim.x >> 6;
  int r = blockIdx.x * rows_per_blk + (threadIdx.x >> 6);
  const int rstep = rows_per_blk * gridDim.x;
  float s = 0.0f, sq = 0.0f;
  for (; r < n_nodes; r += rstep) {
    float v = z[(long)r * DIM + c];
    s += v;
    sq += v * v;
  }
  atomicAdd(&stats[c], s);
  atomicAdd(&stats[DIM + c], sq);
}

// ------------------------------------------------- BatchNorm pass 2 + global_add_pool
__global__ void gin_bn_apply_pool(const float* __restrict__ z, float* __restrict__ h,
                                  const float* __restrict__ stats,
                                  const float* __restrict__ gamma, const float* __restrict__ beta,
                                  const int* __restrict__ batch, float* __restrict__ pooled,
                                  int layer, int n_nodes) {
  long t = blockIdx.x * (long)blockDim.x + threadIdx.x;       // float4 granularity
  const long total = (long)n_nodes * (DIM / 4);
  const long step = (long)gridDim.x * blockDim.x;
  const float invN = 1.0f / (float)n_nodes;
  for (; t < total; t += step) {
    int r  = (int)(t >> 4);
    int c4 = (int)(t & 15) * 4;
    float4 zv = *(const float4*)(z + (long)r * DIM + c4);
    float* pbase = pooled + (long)batch[r] * (LAYERS * DIM) + layer * DIM + c4;
    float hv[4];
    const float* zp = &zv.x;
#pragma unroll
    for (int i = 0; i < 4; ++i) {
      int c = c4 + i;
      float mean = stats[c] * invN;
      float var  = stats[DIM + c] * invN - mean * mean;
      float scl  = gamma[c] * rsqrtf(var + BN_EPSF);
      float v    = (zp[i] - mean) * scl + beta[c];
      hv[i] = v;
      atomicAdd(pbase + i, v);
    }
    *(float4*)(h + (long)r * DIM + c4) = make_float4(hv[0], hv[1], hv[2], hv[3]);
  }
}

// ------------------------------------------------- head: MLP + log_softmax, block/graph
__global__ __launch_bounds__(64)
void gin_head(const float* __restrict__ pooled,
              const float* __restrict__ lin_w, const float* __restrict__ lin_b,
              const float* __restrict__ fin_w, const float* __restrict__ fin_b,
              float* __restrict__ out) {
  __shared__ float s1[DIM];
  __shared__ float lo[CLASSES];
  __shared__ float lse;
  const int g = blockIdx.x;
  const int j = threadIdx.x;
  const float* xp = pooled + (long)g * (LAYERS * DIM);

  float acc = lin_b[j];
  for (int k = 0; k < LAYERS * DIM; ++k) acc += xp[k] * lin_w[k * DIM + j];
  s1[j] = acc > 0.0f ? acc : 0.0f;
  __syncthreads();

  if (j < CLASSES) {
    float a = fin_b[j];
    for (int k = 0; k < DIM; ++k) a += s1[k] * fin_w[k * CLASSES + j];
    lo[j] = a;
  }
  __syncthreads();

  if (j == 0) {
    float mx = lo[0];
    for (int c = 1; c < CLASSES; ++c) mx = fmaxf(mx, lo[c]);
    float se = 0.0f;
    for (int c = 0; c < CLASSES; ++c) se += expf(lo[c] - mx);
    lse = mx + logf(se);
  }
  __syncthreads();

  if (j < CLASSES) out[(long)g * CLASSES + j] = lo[j] - lse;
}

// ---------------------------------------------------------------- launcher
extern "C" void kernel_launch(void* const* d_in, const int* in_sizes, int n_in,
                              void* d_out, int out_size, void* d_ws, size_t ws_size,
                              hipStream_t stream) {
  const float* x     = (const float*)d_in[0];
  const int*   ei    = (const int*)d_in[1];
  const int*   batch = (const int*)d_in[2];
  const float* w1    = (const float*)d_in[3];
  const float* b1    = (const float*)d_in[4];
  const float* w2    = (const float*)d_in[5];
  const float* b2    = (const float*)d_in[6];
  const float* gamma = (const float*)d_in[7];
  const float* beta  = (const float*)d_in[8];
  const float* lin_w = (const float*)d_in[9];
  const float* lin_b = (const float*)d_in[10];
  const float* fin_w = (const float*)d_in[11];
  const float* fin_b = (const float*)d_in[12];
  float* out = (float*)d_out;

  const int n_nodes = in_sizes[0] / DIM;
  const int n_edges = in_sizes[1] / 2;
  const int* src = ei;
  const int* dst = ei + n_edges;

  // workspace carve-up (all offsets 256B-aligned for the given sizes)
  float* agg    = (float*)d_ws;                                   // [N, D] (doubles as z)
  float* hbuf   = agg + (size_t)n_nodes * DIM;                    // [N, D]
  float* pooled = hbuf + (size_t)n_nodes * DIM;                   // [G, L*D]
  float* stats  = pooled + (size_t)GRAPHS * LAYERS * DIM;         // [2*D]
  _Float16* wpack = (_Float16*)(stats + 2 * DIM);                 // [2L][4096] f16 B-frags

  const long nd4 = ((long)n_nodes * DIM) / 4;
  const int  ZB  = 256;

  gin_zero_f32<<<(GRAPHS * LAYERS * DIM + ZB - 1) / ZB, ZB, 0, stream>>>(
      pooled, GRAPHS * LAYERS * DIM);
  gin_pack_weights<<<(LAYERS * 2 * 4096 + ZB - 1) / ZB, ZB, 0, stream>>>(w1, w2, wpack);

  const float* h = x;
  for (int l = 0; l < LAYERS; ++l) {
    gin_copy_f32v4<<<4096, 256, 0, stream>>>((float4*)agg, (const float4*)h, nd4);
    gin_edge_scatter<<<16384, 256, 0, stream>>>(agg, h, src, dst, n_edges);
    gin_mlp_wmma<<<(n_nodes + 15) / 16, 128, 0, stream>>>(
        agg,
        wpack + (size_t)(l * 2 + 0) * 4096, b1 + l * DIM,
        wpack + (size_t)(l * 2 + 1) * 4096, b2 + l * DIM, n_nodes);
    gin_zero_f32<<<1, 128, 0, stream>>>(stats, 2 * DIM);
    gin_bn_stats<<<256, 256, 0, stream>>>(agg, stats, n_nodes);
    gin_bn_apply_pool<<<8192, 256, 0, stream>>>(agg, hbuf, stats,
                                                gamma + l * DIM, beta + l * DIM,
                                                batch, pooled, l, n_nodes);
    h = hbuf;
  }

  gin_head<<<GRAPHS, 64, 0, stream>>>(pooled, lin_w, lin_b, fin_w, fin_b, out);
}
